// EncoderR_44813688766727
// MI455X (gfx1250) — matrix operand order repaired and checked by reading, not compiled, not run on hardware
//
#include <hip/hip_runtime.h>

// ---------------------------------------------------------------------------
// GraFITi-style encoder forward for MI455X (gfx1250, wave32).
// All matmuls -> v_wmma_f32_16x16x32_bf16 (fp32 accumulate, bf16 operands).
// GEMM: one wave computes a 32x32 tile (4 WMMAs / K-step, A+B frag reuse);
// interior tiles use global_load_b128 -> v_cvt_pk_bf16_f32 -> ds_store_b64,
// fragments come back as ds_load_b128 pairs.
// ---------------------------------------------------------------------------

typedef __bf16 bf16;
typedef bf16  v16bf __attribute__((ext_vector_type(16)));
typedef float v8f   __attribute__((ext_vector_type(8)));

#define NB   16      // batch
#define NKC  128     // embedding dim
#define NH   4       // heads
#define HD   32      // head dim
#define SEQ  128     // time steps
#define NCH  42      // channels (41 + 1 extra)

union FragU { v16bf v; uint4 q[2]; };

__device__ __forceinline__ unsigned pk2(float x, float y) {
  unsigned short a = __builtin_bit_cast(unsigned short, (bf16)x);
  unsigned short b = __builtin_bit_cast(unsigned short, (bf16)y);
  return (unsigned)a | ((unsigned)b << 16);
}

// ---------------------------------------------------------------------------
// Generic bf16 WMMA GEMM:
//   D[b,h] = rowscale * ( addsrc + act( alpha * A @ op(B) + bias ) )
// ---------------------------------------------------------------------------
__global__ __launch_bounds__(32)
void gemm_wmma_bf16(
    const float* __restrict__ A, long sAb, long sAh, int lda,
    const float* __restrict__ Bm, long sBb, long sBh, int ldb, int transB,
    float* __restrict__ D, long sDb, long sDh, int ldd,
    const float* __restrict__ bias,
    const float* __restrict__ addsrc, long sCb, long sCh, int ldc,
    const float* __restrict__ rowscale, long sRb,
    int M, int N, int K, int H, float alpha, int do_relu)
{
  int z = blockIdx.z; int b = z / H; int h = z - b * H;
  A  += (long)b * sAb + (long)h * sAh;
  Bm += (long)b * sBb + (long)h * sBh;
  D  += (long)b * sDb + (long)h * sDh;
  if (addsrc)   addsrc   += (long)b * sCb + (long)h * sCh;
  if (rowscale) rowscale += (long)b * sRb;

  int tm = blockIdx.y * 32, tn = blockIdx.x * 32;
  __shared__ bf16 lA [32][40];   // [m][k]  rows padded to 80B (16B-aligned chunks)
  __shared__ bf16 lBt[32][40];   // [n][k]  n-major regardless of transB
  int lane = threadIdx.x;
  v8f acc[4]; acc[0] = {}; acc[1] = {}; acc[2] = {}; acc[3] = {};
  const bool alignOK = ((lda & 3) == 0) && (!transB || ((ldb & 3) == 0));

  for (int k0 = 0; k0 < K; k0 += 32) {
    bool fast = alignOK && (tm + 32 <= M) && (tn + 32 <= N) && (k0 + 32 <= K);
    if (fast) {
      // ---- A: 32x32 via float4 (8 per lane) ----
      #pragma unroll
      for (int t = 0; t < 8; ++t) {
        int i = t * 32 + lane; int r = i >> 3, c4 = (i & 7) * 4;
        const float4 v = *(const float4*)(A + (long)(tm + r) * lda + k0 + c4);
        unsigned long long pk = (unsigned long long)pk2(v.x, v.y)
                              | ((unsigned long long)pk2(v.z, v.w) << 32);
        *(unsigned long long*)&lA[r][c4] = pk;
      }
      // ---- B: 32x32 into n-major LDS; lane owns column n = lane ----
      int n = lane;
      if (!transB) {
        #pragma unroll
        for (int t = 0; t < 8; ++t) {
          int kk = k0 + t * 4;
          float x0 = Bm[(long)(kk + 0) * ldb + tn + n];
          float x1 = Bm[(long)(kk + 1) * ldb + tn + n];
          float x2 = Bm[(long)(kk + 2) * ldb + tn + n];
          float x3 = Bm[(long)(kk + 3) * ldb + tn + n];
          unsigned long long pk = (unsigned long long)pk2(x0, x1)
                                | ((unsigned long long)pk2(x2, x3) << 32);
          *(unsigned long long*)&lBt[n][t * 4] = pk;
        }
      } else {
        #pragma unroll
        for (int t = 0; t < 8; ++t) {
          const float4 v = *(const float4*)(Bm + (long)(tn + n) * ldb + k0 + t * 4);
          unsigned long long pk = (unsigned long long)pk2(v.x, v.y)
                                | ((unsigned long long)pk2(v.z, v.w) << 32);
          *(unsigned long long*)&lBt[n][t * 4] = pk;
        }
      }
      if (k0 + 32 < K)   // prefetch next K-slab of A (global_prefetch_b8)
        __builtin_prefetch(A + (long)(tm + lane) * lda + k0 + 32, 0, 0);
    } else {
      // ---- guarded scalar path (boundary tiles only) ----
      #pragma unroll 1
      for (int t = 0; t < 32; ++t) {
        int gm = tm + t, gk = k0 + lane;
        float v = (gm < M && gk < K) ? A[(long)gm * lda + gk] : 0.f;
        lA[t][lane] = (bf16)v;
      }
      #pragma unroll 1
      for (int t = 0; t < 32; ++t) {
        int gk = k0 + lane, gn = tn + t;
        float v = 0.f;
        if (gk < K && gn < N)
          v = transB ? Bm[(long)gn * ldb + gk] : Bm[(long)gk * ldb + gn];
        lBt[t][lane] = (bf16)v;
      }
    }
    __syncthreads();
    // ---- fragments (ISA 7.12.2): A and B chunks {half*8, 16+half*8} ----
    int half = lane >> 4, r = lane & 15;
    FragU af0, af1, bf0, bf1;
    af0.q[0] = *(const uint4*)&lA[r][half * 8];
    af0.q[1] = *(const uint4*)&lA[r][16 + half * 8];
    af1.q[0] = *(const uint4*)&lA[16 + r][half * 8];
    af1.q[1] = *(const uint4*)&lA[16 + r][16 + half * 8];
    bf0.q[0] = *(const uint4*)&lBt[r][half * 8];
    bf0.q[1] = *(const uint4*)&lBt[r][16 + half * 8];
    bf1.q[0] = *(const uint4*)&lBt[16 + r][half * 8];
    bf1.q[1] = *(const uint4*)&lBt[16 + r][16 + half * 8];
    acc[0] = __builtin_amdgcn_wmma_f32_16x16x32_bf16(false, af0.v, false, bf0.v,
                                                     (short)0, acc[0], false, false);
    acc[1] = __builtin_amdgcn_wmma_f32_16x16x32_bf16(false, af0.v, false, bf1.v,
                                                     (short)0, acc[1], false, false);
    acc[2] = __builtin_amdgcn_wmma_f32_16x16x32_bf16(false, af1.v, false, bf0.v,
                                                     (short)0, acc[2], false, false);
    acc[3] = __builtin_amdgcn_wmma_f32_16x16x32_bf16(false, af1.v, false, bf1.v,
                                                     (short)0, acc[3], false, false);
    __syncthreads();
  }

  // ---- epilogue: C/D layout -> lane(half,n) covers M = 8*half+vi, N = n ----
  int half = lane >> 4, n = lane & 15;
  #pragma unroll
  for (int my = 0; my < 2; ++my) {
    #pragma unroll
    for (int x = 0; x < 2; ++x) {
      int gn = tn + x * 16 + n;
      if (gn < N) {
        #pragma unroll
        for (int vi = 0; vi < 8; ++vi) {
          int gm = tm + my * 16 + 8 * half + vi;
          if (gm < M) {
            float val = alpha * acc[my * 2 + x][vi];
            if (bias)     val += bias[gn];
            if (do_relu)  val = fmaxf(val, 0.f);
            if (addsrc)   val += addsrc[(long)gm * ldc + gn];
            if (rowscale) val *= rowscale[gm];
            D[(long)gm * ldd + gn] = val;
          }
        }
      }
    }
  }
}

// ---------------------------------------------------------------------------
// Ragged pack: per-batch prefix scan over the 128x42 mask, compact into FL
// slots, zero-pad, also emits target_U_/target_mask_ directly into d_out.
// ---------------------------------------------------------------------------
__global__ __launch_bounds__(256)
void pack_kernel(const float* __restrict__ value, const float* __restrict__ mask,
                 const float* __restrict__ tval,  const float* __restrict__ tmask,
                 const float* __restrict__ extra, const float* __restrict__ textra,
                 int FL, int* __restrict__ Tinds, int* __restrict__ Cinds,
                 float* __restrict__ mkp, float* __restrict__ Uin,
                 float* __restrict__ outTU, float* __restrict__ outTM)
{
  int b = blockIdx.x, tid = threadIdx.x;
  const int PER = (SEQ * NCH) / 256;   // 5376/256 = 21
  __shared__ int scan[256];
  __shared__ int total;
  int base = tid * PER;
  int cnt = 0;
  for (int j = 0; j < PER; ++j) {
    int f = base + j; int t = f / NCH, c = f % NCH;
    float mv = (c < NCH - 1) ? mask[((long)b * SEQ + t) * (NCH - 1) + c] : 1.f;
    if (mv > 0.f) cnt++;
  }
  scan[tid] = cnt; __syncthreads();
  for (int off = 1; off < 256; off <<= 1) {
    int v = (tid >= off) ? scan[tid - off] : 0;
    __syncthreads();
    scan[tid] += v;
    __syncthreads();
  }
  if (tid == 255) total = scan[255];
  int pos = scan[tid] - cnt;           // exclusive prefix
  __syncthreads();
  for (int j = 0; j < PER; ++j) {
    int f = base + j; int t = f / NCH, c = f % NCH;
    long vi = ((long)b * SEQ + t) * (NCH - 1) + c;
    float mv = (c < NCH - 1) ? mask[vi] : 1.f;
    if (mv > 0.f) {
      long o = (long)b * FL + pos;
      Tinds[o] = t; Cinds[o] = c; mkp[o] = 1.f;
      float uv = (c < NCH - 1) ? value[vi] : extra[(long)b * SEQ + t];
      float tu = (c < NCH - 1) ? tval[vi]  : textra[(long)b * SEQ + t];
      float tm = (c < NCH - 1) ? tmask[vi] : 0.f;
      Uin[2 * o] = uv; Uin[2 * o + 1] = tm;   // U_ind = 1 - mk + tmk = tmk here
      outTU[o] = tu; outTM[o] = tm;
      pos++;
    }
  }
  int tot = total;
  for (int pp = tot + tid; pp < FL; pp += 256) {
    long o = (long)b * FL + pp;
    Tinds[o] = 0; Cinds[o] = 0; mkp[o] = 0.f;
    Uin[2 * o] = 0.f; Uin[2 * o + 1] = 1.f;   // 1 - 0 + 0
    outTU[o] = 0.f; outTM[o] = 0.f;
  }
}

// ---------------------------------------------------------------------------
// Masked softmax over rows of stride ldr, length L.
// mask(k) = (inds[b,k]==qi) && mk[b,k]>0 (mode 1); mode 0 = no mask.
// ---------------------------------------------------------------------------
__global__ __launch_bounds__(256)
void softmax_kernel(float* __restrict__ att, int Mq, int L, int ldr, int H,
                    const int* __restrict__ inds, const float* __restrict__ mk,
                    int maskMode, int FLs)
{
  int z = blockIdx.z; int b = z / H;
  int qi = blockIdx.x;
  float* row = att + ((long)z * Mq + qi) * ldr;
  const int*   ind = maskMode ? inds + (long)b * FLs : nullptr;
  const float* mm  = maskMode ? mk   + (long)b * FLs : nullptr;
  __shared__ float red[256];
  int tid = threadIdx.x;

  float mx = -3.4e38f;
  for (int k = tid; k < L; k += 256) {
    float a = row[k];
    if (maskMode && !(ind[k] == qi && mm[k] > 0.f)) a = -1e10f;
    mx = fmaxf(mx, a);
  }
  red[tid] = mx; __syncthreads();
  for (int s = 128; s > 0; s >>= 1) { if (tid < s) red[tid] = fmaxf(red[tid], red[tid + s]); __syncthreads(); }
  mx = red[0]; __syncthreads();

  float sm = 0.f;
  for (int k = tid; k < L; k += 256) {
    float a = row[k];
    if (maskMode && !(ind[k] == qi && mm[k] > 0.f)) a = -1e10f;
    sm += __expf(a - mx);
  }
  red[tid] = sm; __syncthreads();
  for (int s = 128; s > 0; s >>= 1) { if (tid < s) red[tid] += red[tid + s]; __syncthreads(); }
  float inv = 1.f / red[0];

  for (int k = tid; k < L; k += 256) {
    float a = row[k];
    if (maskMode && !(ind[k] == qi && mm[k] > 0.f)) a = -1e10f;
    row[k] = __expf(a - mx) * inv;
  }
}

// ------------------------- small elementwise kernels -----------------------
__global__ void init_T_kernel(const float* __restrict__ x, const float* __restrict__ w,
                              const float* __restrict__ b, float* __restrict__ T, size_t total)
{
  size_t i = (size_t)blockIdx.x * blockDim.x + threadIdx.x;
  if (i >= total) return;
  int n = (int)(i & (NKC - 1)); size_t bs = i >> 7;
  T[i] = sinf(w[n] * x[bs] + b[n]);
}

__global__ void init_C_kernel(const float* __restrict__ w, const float* __restrict__ b,
                              float* __restrict__ C, size_t total)
{
  size_t i = (size_t)blockIdx.x * blockDim.x + threadIdx.x;
  if (i >= total) return;
  int n = (int)(i & (NKC - 1)); int c = (int)((i >> 7) % NCH);
  C[i] = fmaxf(0.f, w[(size_t)c * NKC + n] + b[n]);   // one-hot @ W == row c of W
}

__global__ void gather_kernel(const float* __restrict__ src, const int* __restrict__ inds,
                              float* __restrict__ dst, int FL, int rowsSrc, size_t total)
{
  size_t i = (size_t)blockIdx.x * blockDim.x + threadIdx.x;
  if (i >= total) return;
  int n = (int)(i & (NKC - 1)); size_t bp = i >> 7;
  size_t b = bp / (size_t)FL; size_t pos = bp - b * (size_t)FL;
  int id = inds[b * (size_t)FL + pos];
  dst[i] = src[((size_t)b * rowsSrc + id) * NKC + n];
}

__global__ void edge_update_kernel(float* __restrict__ U, const float* __restrict__ E,
                                   const float* __restrict__ mk, size_t total)
{
  size_t i = (size_t)blockIdx.x * blockDim.x + threadIdx.x;
  if (i >= total) return;
  size_t row = i >> 7;
  U[i] = fmaxf(0.f, U[i] + E[i]) * mk[row];
}

__global__ void output_kernel(const float* __restrict__ U, const float* __restrict__ kt,
                              const float* __restrict__ kc, const float* __restrict__ w,
                              const float* __restrict__ bias, float* __restrict__ out,
                              size_t total)
{
  size_t i = (size_t)blockIdx.x * blockDim.x + threadIdx.x;
  if (i >= total) return;
  const float* u = U  + i * NKC;
  const float* a = kt + i * NKC;
  const float* c = kc + i * NKC;
  float s = bias[0];
  for (int n = 0; n < NKC; ++n) s += u[n] * w[n];
  for (int n = 0; n < NKC; ++n) s += a[n] * w[NKC + n];
  for (int n = 0; n < NKC; ++n) s += c[n] * w[2 * NKC + n];
  out[i] = s;
}

// ---------------------------------------------------------------------------
// Host-side orchestration
// ---------------------------------------------------------------------------
extern "C" void kernel_launch(void* const* d_in, const int* in_sizes, int n_in,
                              void* d_out, int out_size, void* d_ws, size_t ws_size,
                              hipStream_t stream)
{
  const float* ctx    = (const float*)d_in[0];
  const float* value  = (const float*)d_in[1];
  const float* mask   = (const float*)d_in[2];
  const float* tval   = (const float*)d_in[3];
  const float* tmask  = (const float*)d_in[4];
  const float* extra  = (const float*)d_in[5];
  const float* textra = (const float*)d_in[6];
  // d_in[7] = full_len (device scalar) — FL derived on host from out_size:
  const int FL  = out_size / (3 * NB);
  const int FLp = (FL + 3) & ~3;        // att row stride, 16B-aligned rows

  // params in jax pytree-flatten order (dict keys sorted; {b,w} per linear;
  // {k,o,q,v} per MAB; layers: {cc, ct, edge, tc}).
  int pi = 8;
  auto nxt = [&]() { return (const float*)d_in[pi++]; };
  const float* chan_b = nxt(); const float* chan_w = nxt();
  const float* e0_b   = nxt(); const float* e0_w   = nxt();
  struct Mab { const float *kb,*kw,*ob,*ow,*qb,*qw,*vb,*vw; };
  struct Layer { Mab cc, ct; const float *eb, *ew; Mab tc; } L[3];
  auto getMab = [&](Mab& m) {
    m.kb = nxt(); m.kw = nxt(); m.ob = nxt(); m.ow = nxt();
    m.qb = nxt(); m.qw = nxt(); m.vb = nxt(); m.vw = nxt();
  };
  for (int l = 0; l < 3; ++l) {
    getMab(L[l].cc); getMab(L[l].ct);
    L[l].eb = nxt(); L[l].ew = nxt();
    getMab(L[l].tc);
  }
  const float* out_b = nxt(); const float* out_w = nxt();
  const float* tw_b  = nxt(); const float* tw_w  = nxt();

  // --------------------------- workspace carve ----------------------------
  const size_t nBF = (size_t)NB * FL;
  float* p = (float*)d_ws;
  int*   Tinds = (int*)p;  p += nBF;
  int*   Cinds = (int*)p;  p += nBF;
  float* mkp = p;          p += nBF;
  float* Uin = p;          p += 2 * nBF;
  float* U   = p;          p += nBF * NKC;
  float* T0  = p;          p += (size_t)NB * SEQ * NKC;
  float* T1  = p;          p += (size_t)NB * SEQ * NKC;
  float* Cb  = p;          p += (size_t)NB * NCH * NKC;
  float* C2  = p;          p += (size_t)NB * NCH * NKC;
  float* kt  = p;          p += nBF * NKC;
  float* kc  = p;          p += nBF * NKC;
  float* kKV = p;          p += nBF * NKC;
  float* vKV = p;          p += nBF * NKC;
  float* kS  = p;          p += (size_t)NB * NCH * NKC;
  float* vS  = p;          p += (size_t)NB * NCH * NKC;
  float* qS  = p;          p += (size_t)NB * SEQ * NKC;
  float* oS  = p;          p += (size_t)NB * SEQ * NKC;
  float* E   = p;          p += nBF * NKC;
  float* att = p;          /* NB*NH*SEQ*FLp floats */

  float* outMain = (float*)d_out;
  float* outTU   = outMain + nBF;
  float* outTM   = outMain + 2 * nBF;

  auto gemm = [&](const float* A, long sAb, long sAh, int lda,
                  const float* Bm, long sBb, long sBh, int ldb, int tB,
                  float* D, long sDb, long sDh, int ldd,
                  const float* bias,
                  const float* add, long sCb, long sCh, int ldc,
                  const float* rs, long sRb,
                  int M, int N, int K, int H, float alpha, int relu)
  {
    dim3 g((unsigned)((N + 31) / 32), (unsigned)((M + 31) / 32), (unsigned)(NB * H));
    gemm_wmma_bf16<<<g, 32, 0, stream>>>(A, sAb, sAh, lda, Bm, sBb, sBh, ldb, tB,
        D, sDb, sDh, ldd, bias, add, sCb, sCh, ldc, rs, sRb, M, N, K, H, alpha, relu);
  };

  const float ascale = 0.088388347648318447f;   // 1/sqrt(128)

  // Residual MHA + residual relu-linear (GraFITi MAB2).
  // K input = [K1 | K2] (K2 may be null): split-K accumulation into kbuf/vbuf.
  auto run_mab = [&](const Mab& m, const float* Q, int Mq,
                     const float* K1, const float* K2, int Lk,
                     float* kbuf, float* vbuf, float* Dout,
                     const int* mInds, int mMode)
  {
    long sQ = (long)Mq * NKC, sK = (long)Lk * NKC;
    long sAttH = (long)Mq * FLp, sAttB = (long)NH * Mq * FLp;
    // q / k / v projections
    gemm(Q, sQ, 0, NKC, m.qw, 0, 0, NKC, 0, qS, sQ, 0, NKC, m.qb,
         nullptr, 0, 0, 0, nullptr, 0, Mq, NKC, NKC, 1, 1.f, 0);
    gemm(K1, sK, 0, NKC, m.kw, 0, 0, NKC, 0, kbuf, sK, 0, NKC, m.kb,
         nullptr, 0, 0, 0, nullptr, 0, Lk, NKC, NKC, 1, 1.f, 0);
    if (K2)
      gemm(K2, sK, 0, NKC, m.kw + NKC * NKC, 0, 0, NKC, 0, kbuf, sK, 0, NKC, nullptr,
           kbuf, sK, 0, NKC, nullptr, 0, Lk, NKC, NKC, 1, 1.f, 0);
    gemm(K1, sK, 0, NKC, m.vw, 0, 0, NKC, 0, vbuf, sK, 0, NKC, m.vb,
         nullptr, 0, 0, 0, nullptr, 0, Lk, NKC, NKC, 1, 1.f, 0);
    if (K2)
      gemm(K2, sK, 0, NKC, m.vw + NKC * NKC, 0, 0, NKC, 0, vbuf, sK, 0, NKC, nullptr,
           vbuf, sK, 0, NKC, nullptr, 0, Lk, NKC, NKC, 1, 1.f, 0);
    // scores: att[b,h] = q_h @ k_h^T / sqrt(128)   (B transposed)
    gemm(qS, sQ, HD, NKC, kbuf, sK, HD, NKC, 1,
         att, sAttB, sAttH, FLp,
         nullptr, nullptr, 0, 0, 0, nullptr, 0, Mq, Lk, HD, NH, ascale, 0);
    softmax_kernel<<<dim3((unsigned)Mq, 1, NB * NH), 256, 0, stream>>>(
        att, Mq, Lk, FLp, NH, mInds, mMode ? mkp : nullptr, mMode, FL);
    // o = q_h + a @ v_h
    gemm(att, sAttB, sAttH, FLp, vbuf, sK, HD, NKC, 0,
         oS, sQ, HD, NKC, nullptr, qS, sQ, HD, NKC, nullptr, 0,
         Mq, HD, Lk, NH, 1.f, 0);
    // out = o + relu(o @ Wo + bo)
    gemm(oS, sQ, 0, NKC, m.ow, 0, 0, NKC, 0, Dout, sQ, 0, NKC, m.ob,
         oS, sQ, 0, NKC, nullptr, 0, Mq, NKC, NKC, 1, 1.f, 1);
  };

  // ------------------------------- pipeline -------------------------------
  pack_kernel<<<NB, 256, 0, stream>>>(value, mask, tval, tmask, extra, textra,
                                      FL, Tinds, Cinds, mkp, Uin, outTU, outTM);
  {
    size_t tT = (size_t)NB * SEQ * NKC;
    init_T_kernel<<<(unsigned)((tT + 255) / 256), 256, 0, stream>>>(ctx, tw_w, tw_b, T0, tT);
    size_t tC = (size_t)NB * NCH * NKC;
    init_C_kernel<<<(unsigned)((tC + 255) / 256), 256, 0, stream>>>(chan_w, chan_b, Cb, tC);
  }
  // U_ = relu(edge_init([U_vals, U_ind])) * mk_
  gemm(Uin, (long)FL * 2, 0, 2, e0_w, 0, 0, NKC, 0, U, (long)FL * NKC, 0, NKC, e0_b,
       nullptr, 0, 0, 0, mkp, (long)FL, FL, NKC, 2, 1, 1.f, 1);

  float* Tcur = T0; float* Toth = T1;
  const size_t gtot = (size_t)NB * FL * NKC;
  const unsigned ggrid = (unsigned)((gtot + 255) / 256);

  for (int l = 0; l < 3; ++l) {
    gather_kernel<<<ggrid, 256, 0, stream>>>(Tcur, Tinds, kt, FL, SEQ, gtot);
    gather_kernel<<<ggrid, 256, 0, stream>>>(Cb,   Cinds, kc, FL, NCH, gtot);
    // C__ = MAB_ct(C_, [k_t, U_], C_mask)
    run_mab(L[l].ct, Cb, NCH, kt, U, FL, kKV, vKV, C2, Cinds, 1);
    // T__ = MAB_tc(T_, [k_c, U_], T_mask)
    run_mab(L[l].tc, Tcur, SEQ, kc, U, FL, kKV, vKV, Toth, Tinds, 1);
    // U_ = relu(U_ + edge([U_, k_t, k_c])) * mk_   (split-K accumulation)
    gemm(U,  (long)FL * NKC, 0, NKC, L[l].ew,               0, 0, NKC, 0,
         E, (long)FL * NKC, 0, NKC, L[l].eb, nullptr, 0, 0, 0, nullptr, 0,
         FL, NKC, NKC, 1, 1.f, 0);
    gemm(kt, (long)FL * NKC, 0, NKC, L[l].ew + NKC * NKC,   0, 0, NKC, 0,
         E, (long)FL * NKC, 0, NKC, nullptr, E, (long)FL * NKC, 0, NKC, nullptr, 0,
         FL, NKC, NKC, 1, 1.f, 0);
    gemm(kc, (long)FL * NKC, 0, NKC, L[l].ew + 2 * NKC * NKC, 0, 0, NKC, 0,
         E, (long)FL * NKC, 0, NKC, nullptr, E, (long)FL * NKC, 0, NKC, nullptr, 0,
         FL, NKC, NKC, 1, 1.f, 0);
    edge_update_kernel<<<ggrid, 256, 0, stream>>>(U, E, mkp, gtot);
    // C_ = MAB_cc(C__, C__)   (no mask)
    run_mab(L[l].cc, C2, NCH, C2, nullptr, NCH, kS, vS, Cb, nullptr, 0);
    float* tsw = Tcur; Tcur = Toth; Toth = tsw;   // T_ = T__
  }

  gather_kernel<<<ggrid, 256, 0, stream>>>(Tcur, Tinds, kt, FL, SEQ, gtot);
  gather_kernel<<<ggrid, 256, 0, stream>>>(Cb,   Cinds, kc, FL, NCH, gtot);
  {
    size_t rows = nBF;
    output_kernel<<<(unsigned)((rows + 255) / 256), 256, 0, stream>>>(
        U, kt, kc, out_w, out_b, outMain, rows);
  }
}
